// Transformer_21852793602054
// MI455X (gfx1250) — compile-verified, matrix operand order checked by reading, not run on hardware
//
#include <hip/hip_runtime.h>
#include <math.h>

// ---------------------------------------------------------------------------
// CDNA5 / gfx1250 implementation.  All heavy math goes through
// v_wmma_f32_16x16x32_bf16 (wave32 WMMA, f32 accumulate).
// ---------------------------------------------------------------------------

typedef __attribute__((ext_vector_type(16))) __bf16 v16bf;
typedef __attribute__((ext_vector_type(8)))  __bf16 v8bf;
typedef __attribute__((ext_vector_type(4)))  __bf16 v4bf;
typedef __attribute__((ext_vector_type(8)))  float  v8f;
typedef __attribute__((ext_vector_type(4)))  float  v4f;

#define B_    64
#define T_    256
#define BT_   16384          // B*T
#define W_    16
#define GLOVE_ 300
#define CHOUT_ 250
#define EMB_   1604
#define HWD_   550
#define K_     512
#define H_     8
#define HK_    4096          // HEADS*K
#define XDIM_  1354          // GLOVE+ELMO+POS

// GEMM tiling: 128x64 workgroup tile, 8 waves, each wave owns 32x32
// (four 16x16 WMMA accumulators).  K-step 32.
#define TM 128
#define TN 64
#define TK 32
#define KP 40                // padded k-stride (bf16 elems) -> 80B rows, 16B-aligned

__device__ __forceinline__ float act_apply(float v, int act) {
  switch (act) {
    case 1:  return v > 0.f ? v : 0.f;                 // relu
    case 2:  return tanhf(v);                          // tanh
    case 3:  return 1.f / (1.f + __expf(-v));          // sigmoid
    default: return v;
  }
}

__device__ __forceinline__ v16bf frag_load(const __bf16* p) {
  // two contiguous 16B runs: elements [0..7] and [16..23] of a KP-strided row
  v8bf lo = *(const v8bf*)p;
  v8bf hi = *(const v8bf*)(p + 16);
  return __builtin_shufflevector(lo, hi, 0, 1, 2, 3, 4, 5, 6, 7,
                                         8, 9, 10, 11, 12, 13, 14, 15);
}

// ---------------------------------------------------------------------------
// Batched, strided GEMM:  C = act(alpha * A@B + bias)
//   A: M x Kd (row major, lda)
//   B: btrans==0 -> Kd x N (ldb);  btrans==1 -> N x Kd (ldb)
//   C: M x N  (row major, ldc)
// batch z decomposes as z = b0*bdiv + b1 with two-level strides per operand.
// A and B tiles staged into LDS as bf16; B is staged TRANSPOSED (Bt[n][k]) so
// both fragment loads are 2x ds_load_b128.  Inner product: 4x
// v_wmma_f32_16x16x32_bf16 per wave per k-step.
//
// MODE 0: general (bounds-checked, runtime btrans)
// MODE 1: M%TM==0, N%TN==0, Kd%TK==0, lda%4==0, ldb%4==0, B row-major KdxN
// MODE 2: same regularity, B transposed NxKd
// ---------------------------------------------------------------------------
template <int MODE>
__global__ __launch_bounds__(256)
void gemm_wmma_bf16(const float* __restrict__ A, const float* __restrict__ Bm,
                    const float* __restrict__ bias, float* __restrict__ C,
                    int M, int N, int Kd,
                    long long lda, long long ldb, long long ldc,
                    int bdiv,
                    long long sA0, long long sA1,
                    long long sB0, long long sB1,
                    long long sC0, long long sC1,
                    int btrans, int act, float alpha)
{
  __shared__ __bf16 As[TM * KP];   // row-major:  As[m][k]
  __shared__ __bf16 Bt[TN * KP];   // transposed: Bt[n][k]

  const int bz = blockIdx.z;
  const int b0 = bz / bdiv, b1 = bz % bdiv;
  A  += (long long)b0 * sA0 + (long long)b1 * sA1;
  Bm += (long long)b0 * sB0 + (long long)b1 * sB1;
  C  += (long long)b0 * sC0 + (long long)b1 * sC1;

  const int tidx = threadIdx.x;
  const int lane = tidx & 31;
  const int wave = tidx >> 5;
  const int wm   = wave & 3;    // M sub-block (32 rows)
  const int wn   = wave >> 2;   // N sub-block (32 cols)

  const int m0 = blockIdx.y * TM;
  const int n0 = blockIdx.x * TN;

  v8f acc00 = {}, acc01 = {}, acc10 = {}, acc11 = {};

  for (int k0 = 0; k0 < Kd; k0 += TK) {
    // ---------------- stage A tile (TM x TK) ----------------
    if constexpr (MODE != 0) {
      // 4096 elems = 1024 float4 chunks, 4 per thread -> global_load_b128
#pragma unroll
      for (int i = 0; i < 4; ++i) {
        int id = tidx + i * 256;
        int r = id >> 3, c4 = (id & 7) * 4;
        v4f d = *(const v4f*)(A + (long long)(m0 + r) * lda + k0 + c4);
        v4bf p;
        p[0] = (__bf16)d[0]; p[1] = (__bf16)d[1];
        p[2] = (__bf16)d[2]; p[3] = (__bf16)d[3];
        *(v4bf*)(&As[r * KP + c4]) = p;                 // ds_store_b64
      }
      if (k0 + TK < Kd) {
        int r = tidx >> 1, c = (tidx & 1) * 16;
        __builtin_prefetch(&A[(long long)(m0 + r) * lda + k0 + TK + c], 0, 1);
      }
    } else {
      const bool fullA = (k0 + TK <= Kd) && (m0 + TM <= M) && ((lda & 3) == 0);
      if (fullA) {
#pragma unroll
        for (int i = 0; i < 4; ++i) {
          int id = tidx + i * 256;
          int r = id >> 3, c4 = (id & 7) * 4;
          v4f d = *(const v4f*)(A + (long long)(m0 + r) * lda + k0 + c4);
          v4bf p;
          p[0] = (__bf16)d[0]; p[1] = (__bf16)d[1];
          p[2] = (__bf16)d[2]; p[3] = (__bf16)d[3];
          *(v4bf*)(&As[r * KP + c4]) = p;
        }
      } else {
#pragma unroll
        for (int i = 0; i < 16; ++i) {
          int id = tidx + i * 256;
          int r = id >> 5, c = id & 31;
          int gm = m0 + r, gk = k0 + c;
          float v = (gm < M && gk < Kd) ? A[(long long)gm * lda + gk] : 0.f;
          As[r * KP + c] = (__bf16)v;
        }
      }
    }

    // ---------------- stage B tile transposed (TN x TK) ----------------
    if constexpr (MODE == 2) {
      // B is N x Kd: k contiguous -> float4 along k, ds_store_b64
#pragma unroll
      for (int i = 0; i < 2; ++i) {
        int id = tidx + i * 256;
        int n = id >> 3, k4 = (id & 7) * 4;
        v4f d = *(const v4f*)(Bm + (long long)(n0 + n) * ldb + k0 + k4);
        v4bf p;
        p[0] = (__bf16)d[0]; p[1] = (__bf16)d[1];
        p[2] = (__bf16)d[2]; p[3] = (__bf16)d[3];
        *(v4bf*)(&Bt[n * KP + k4]) = p;
      }
    } else if constexpr (MODE == 1) {
      // B is Kd x N: n contiguous -> float4 along n, 4 scalar LDS stores
#pragma unroll
      for (int i = 0; i < 2; ++i) {
        int id = tidx + i * 256;
        int k = id >> 4, n4 = (id & 15) * 4;
        v4f d = *(const v4f*)(Bm + (long long)(k0 + k) * ldb + n0 + n4);
        Bt[(n4 + 0) * KP + k] = (__bf16)d[0];
        Bt[(n4 + 1) * KP + k] = (__bf16)d[1];
        Bt[(n4 + 2) * KP + k] = (__bf16)d[2];
        Bt[(n4 + 3) * KP + k] = (__bf16)d[3];
      }
    } else {
      const bool fullB = (k0 + TK <= Kd) && (n0 + TN <= N) && ((ldb & 3) == 0);
      if (fullB && btrans) {
#pragma unroll
        for (int i = 0; i < 2; ++i) {
          int id = tidx + i * 256;
          int n = id >> 3, k4 = (id & 7) * 4;
          v4f d = *(const v4f*)(Bm + (long long)(n0 + n) * ldb + k0 + k4);
          v4bf p;
          p[0] = (__bf16)d[0]; p[1] = (__bf16)d[1];
          p[2] = (__bf16)d[2]; p[3] = (__bf16)d[3];
          *(v4bf*)(&Bt[n * KP + k4]) = p;
        }
      } else if (fullB) {
#pragma unroll
        for (int i = 0; i < 2; ++i) {
          int id = tidx + i * 256;
          int k = id >> 4, n4 = (id & 15) * 4;
          v4f d = *(const v4f*)(Bm + (long long)(k0 + k) * ldb + n0 + n4);
          Bt[(n4 + 0) * KP + k] = (__bf16)d[0];
          Bt[(n4 + 1) * KP + k] = (__bf16)d[1];
          Bt[(n4 + 2) * KP + k] = (__bf16)d[2];
          Bt[(n4 + 3) * KP + k] = (__bf16)d[3];
        }
      } else {
#pragma unroll
        for (int i = 0; i < 8; ++i) {
          int id = tidx + i * 256;
          int k = id >> 6, n = id & 63;
          int gk = k0 + k, gn = n0 + n;
          float v = 0.f;
          if (gk < Kd && gn < N)
            v = btrans ? Bm[(long long)gn * ldb + gk]
                       : Bm[(long long)gk * ldb + gn];
          Bt[n * KP + k] = (__bf16)v;
        }
      }
    }
    __syncthreads();

    // ---------------- fragments: each 2x ds_load_b128 ----------------
    const int kh = (lane >> 4) * 8;
    const int ar = wm * 32 + (lane & 15);
    const int br = wn * 32 + (lane & 15);
    v16bf fa0 = frag_load(&As[ar * KP + kh]);
    v16bf fa1 = frag_load(&As[(ar + 16) * KP + kh]);
    v16bf fb0 = frag_load(&Bt[br * KP + kh]);
    v16bf fb1 = frag_load(&Bt[(br + 16) * KP + kh]);

    acc00 = __builtin_amdgcn_wmma_f32_16x16x32_bf16(false, fa0, false, fb0,
                                                    (short)0, acc00, false, false);
    acc01 = __builtin_amdgcn_wmma_f32_16x16x32_bf16(false, fa0, false, fb1,
                                                    (short)0, acc01, false, false);
    acc10 = __builtin_amdgcn_wmma_f32_16x16x32_bf16(false, fa1, false, fb0,
                                                    (short)0, acc10, false, false);
    acc11 = __builtin_amdgcn_wmma_f32_16x16x32_bf16(false, fa1, false, fb1,
                                                    (short)0, acc11, false, false);
    __syncthreads();
  }

  // ---- epilogue (C/D layout: n = lane%16, m = vgpr + 8*(lane/16)) ----
  const int mb = m0 + wm * 32 + (lane >> 4) * 8;
  const int nb = n0 + wn * 32 + (lane & 15);
#pragma unroll
  for (int half = 0; half < 2; ++half) {
    const v8f* aL = half ? &acc10 : &acc00;
    const v8f* aR = half ? &acc11 : &acc01;
    const int mh = mb + half * 16;
#pragma unroll
    for (int r = 0; r < 8; ++r) {
      int gm = mh + r;
      if (MODE != 0 || gm < M) {
        float bL = bias ? bias[nb] : 0.f;
        float bR = (bias && (MODE != 0 || nb + 16 < N)) ? bias[nb + 16] : 0.f;
        if (MODE != 0 || nb < N) {
          float v = alpha * (*aL)[r] + bL;
          C[(long long)gm * ldc + nb] = act_apply(v, act);
        }
        if (MODE != 0 || nb + 16 < N) {
          float v = alpha * (*aR)[r] + bR;
          C[(long long)gm * ldc + nb + 16] = act_apply(v, act);
        }
      }
    }
  }
}

// ---------------------------------------------------------------------------
// Character CNN: one workgroup per (b,t); 16x50 embedding tile in LDS,
// 250 output channels (4 conv widths), tanh + max-over-positions.
// ---------------------------------------------------------------------------
__global__ __launch_bounds__(256)
void char_cnn_kernel(const float* __restrict__ emb, const int* __restrict__ seqs,
                     const float* __restrict__ w0, const float* __restrict__ b0,
                     const float* __restrict__ w1, const float* __restrict__ b1,
                     const float* __restrict__ w2, const float* __restrict__ b2,
                     const float* __restrict__ w3, const float* __restrict__ b3,
                     float* __restrict__ out)
{
  __shared__ float ce[W_ * 50];
  const int bt = blockIdx.x;
  for (int i = threadIdx.x; i < W_ * 50; i += 256) {
    int wch = i / 50, e = i % 50;
    int ci = seqs[bt * W_ + wch];
    ce[i] = emb[ci * 50 + e];
  }
  __syncthreads();
  for (int c = threadIdx.x; c < CHOUT_; c += 256) {
    const float* Wp; const float* Bp; int kw, cl;
    if (c < 25)       { Wp = w0; Bp = b0; kw = 1; cl = c; }
    else if (c < 75)  { Wp = w1; Bp = b1; kw = 2; cl = c - 25; }
    else if (c < 150) { Wp = w2; Bp = b2; kw = 3; cl = c - 75; }
    else              { Wp = w3; Bp = b3; kw = 4; cl = c - 150; }
    const float bb = Bp[cl];
    float m = -3.0e38f;
    for (int p = 0; p <= W_ - kw; ++p) {
      float acc = bb;
      for (int i = 0; i < kw; ++i)
        for (int e = 0; e < 50; ++e)
          acc += Wp[(cl * 50 + e) * kw + i] * ce[(p + i) * 50 + e];
      m = fmaxf(m, tanhf(acc));
    }
    out[bt * CHOUT_ + c] = m;
  }
}

// hwin = concat(glove, char_out)  -> (BT, 550)
__global__ void build_hwin_kernel(const float* __restrict__ x,
                                  const float* __restrict__ ce,
                                  float* __restrict__ hwin)
{
  int i = blockIdx.x * blockDim.x + threadIdx.x;
  if (i >= BT_ * HWD_) return;
  int row = i / HWD_, col = i % HWD_;
  hwin[i] = (col < GLOVE_) ? x[(long long)row * XDIM_ + col]
                           : ce[row * CHOUT_ + (col - GLOVE_)];
}

// xcat = concat(highway(h), elmo, pos)   highway: tg*f1 + (1-tg)*h
__global__ void build_xcat_kernel(const float* __restrict__ x,
                                  const float* __restrict__ hwin,
                                  const float* __restrict__ f1,
                                  const float* __restrict__ tg,
                                  float* __restrict__ xcat)
{
  int i = blockIdx.x * blockDim.x + threadIdx.x;
  if (i >= BT_ * EMB_) return;
  int row = i / EMB_, col = i % EMB_;
  float v;
  if (col < HWD_) {
    long long j = (long long)row * HWD_ + col;
    float t = tg[j];
    v = t * f1[j] + (1.f - t) * hwin[j];
  } else {
    v = x[(long long)row * XDIM_ + (col - HWD_ + GLOVE_)];
  }
  xcat[i] = v;
}

// masked softmax over last dim of (B,H,T,T); mask where pad_q && pad_k
__global__ __launch_bounds__(256)
void softmax_mask_kernel(float* __restrict__ sc, const int* __restrict__ pad)
{
  __shared__ float red[256];
  const int row = blockIdx.x;          // b*H*T + h*T + tq
  const int tq  = row & (T_ - 1);
  const int b   = row / (H_ * T_);
  const int lim = T_ - pad[b];
  const bool padq = tq >= lim;
  float* r = sc + (long long)row * T_;
  const int tk = threadIdx.x;
  float v = r[tk];
  if (padq && tk >= lim) v = -3.0e38f;
  red[tk] = v; __syncthreads();
  for (int s = 128; s > 0; s >>= 1) {
    if (tk < s) red[tk] = fmaxf(red[tk], red[tk + s]);
    __syncthreads();
  }
  float mx = red[0]; __syncthreads();
  float e = __expf(v - mx);
  red[tk] = e; __syncthreads();
  for (int s = 128; s > 0; s >>= 1) {
    if (tk < s) red[tk] += red[tk + s];
    __syncthreads();
  }
  r[tk] = e / red[0];
}

// out = LayerNorm(a + y) * g + b   (row width 512)
__global__ __launch_bounds__(256)
void add_ln_kernel(const float* __restrict__ a, const float* __restrict__ y,
                   const float* __restrict__ g, const float* __restrict__ b,
                   float* __restrict__ out)
{
  __shared__ float rs[256];
  __shared__ float rq[256];
  const long long base = (long long)blockIdx.x * K_;
  const int t = threadIdx.x;
  float v0 = a[base + t]       + y[base + t];
  float v1 = a[base + t + 256] + y[base + t + 256];
  rs[t] = v0 + v1;
  rq[t] = v0 * v0 + v1 * v1;
  __syncthreads();
  for (int s = 128; s > 0; s >>= 1) {
    if (t < s) { rs[t] += rs[t + s]; rq[t] += rq[t + s]; }
    __syncthreads();
  }
  float mean = rs[0] * (1.f / K_);
  float var  = rq[0] * (1.f / K_) - mean * mean;
  float inv  = rsqrtf(var + 1e-5f);
  out[base + t]       = (v0 - mean) * inv * g[t]       + b[t];
  out[base + t + 256] = (v1 - mean) * inv * g[t + 256] + b[t + 256];
}

__global__ void mul_kernel(const float* __restrict__ a, const float* __restrict__ b,
                           float* __restrict__ c, int n)
{
  int i = blockIdx.x * blockDim.x + threadIdx.x;
  if (i < n) c[i] = a[i] * b[i];
}

__global__ void copy_kernel(const float* __restrict__ a, float* __restrict__ c, int n)
{
  int i = blockIdx.x * blockDim.x + threadIdx.x;
  if (i < n) c[i] = a[i];
}

// feats(50) @ toprobs(50x2) + bias -> logits + log_softmax
__global__ void final_logits_kernel(const float* __restrict__ feats,
                                    const float* __restrict__ tw,
                                    const float* __restrict__ tb,
                                    float* __restrict__ out_logp,
                                    float* __restrict__ out_logits)
{
  int i = blockIdx.x * blockDim.x + threadIdx.x;
  if (i >= BT_) return;
  float l0 = tb[0], l1 = tb[1];
  for (int f = 0; f < 50; ++f) {
    float fv = feats[i * 50 + f];
    l0 += fv * tw[f * 2 + 0];
    l1 += fv * tw[f * 2 + 1];
  }
  float m = fmaxf(l0, l1);
  float lse = m + logf(__expf(l0 - m) + __expf(l1 - m));
  out_logp[i * 2 + 0]   = l0 - lse;
  out_logp[i * 2 + 1]   = l1 - lse;
  out_logits[i * 2 + 0] = l0;
  out_logits[i * 2 + 1] = l1;
}

// ---------------------------------------------------------------------------
// Host orchestration
// ---------------------------------------------------------------------------
extern "C" void kernel_launch(void* const* d_in, const int* in_sizes, int n_in,
                              void* d_out, int out_size, void* d_ws, size_t ws_size,
                              hipStream_t stream)
{
  (void)in_sizes; (void)n_in; (void)out_size; (void)ws_size;
  auto P = [&](int i) { return (const float*)d_in[i]; };

  // jax tree flatten order (dicts sorted by key; lists in order):
  // blocks[0]: bu,ff1_b,ff1_w,ff2_b,ff2_w,ln1_b,ln1_g,ln2_b,ln2_g,tr_b,tr_w,wk,wq,wu,wv (0..14)
  // blocks[1..3]: bu,ff1_b,ff1_w,ff2_b,ff2_w,ln1_b,ln1_g,ln2_b,ln2_g,wk,wq,wu,wv (13 each)
  // 54 char_emb; 55..62 convs (b,w)x4; 63 features_w; 64 hw_t1_b; 65 hw_t1_w;
  // 66 hw_t2_b; 67 hw_t2_w; 68 toprobs_b; 69 toprobs_w; 70 transform_b; 71 transform_w
  // 72 x; 73 pad_amounts; 74 char_seqs
  const float* char_emb   = P(54);
  const float* conv_b[4]  = { P(55), P(57), P(59), P(61) };
  const float* conv_w[4]  = { P(56), P(58), P(60), P(62) };
  const float* features_w = P(63);
  const float* hw_t1_b = P(64); const float* hw_t1_w = P(65);
  const float* hw_t2_b = P(66); const float* hw_t2_w = P(67);
  const float* toprobs_b = P(68); const float* toprobs_w = P(69);
  const float* transform_b = P(70); const float* transform_w = P(71);
  const float* x = P(72);
  const int* pad  = (const int*)d_in[73];
  const int* seqs = (const int*)d_in[74];

  // workspace layout (floats) — every buffer fully written before read
  float* ws = (float*)d_ws;
  size_t o = 0;
  float* CE   = ws + o; o += (size_t)BT_ * CHOUT_;
  float* HWIN = ws + o; o += (size_t)BT_ * HWD_;
  float* F1   = ws + o; o += (size_t)BT_ * HWD_;
  float* TG   = ws + o; o += (size_t)BT_ * HWD_;
  float* XCAT = ws + o; o += (size_t)BT_ * EMB_;
  float* Qb   = ws + o; o += (size_t)BT_ * HK_;
  float* Kb   = ws + o; o += (size_t)BT_ * HK_;
  float* Vb   = ws + o; o += (size_t)BT_ * HK_;
  float* SC   = ws + o; o += (size_t)B_ * H_ * T_ * T_;
  float* AO   = ws + o; o += (size_t)BT_ * HK_;
  float* Ab   = ws + o; o += (size_t)BT_ * K_;
  float* Yb   = ws + o; o += (size_t)BT_ * K_;
  float* H1   = ws + o; o += (size_t)BT_ * K_;
  float* FF   = ws + o; o += (size_t)BT_ * 2048;
  float* Zb   = ws + o; o += (size_t)BT_ * K_;
  float* PROJ = ws + o; o += (size_t)BT_ * K_;
  float* PZ   = ws + o; o += (size_t)BT_ * K_;
  float* FE   = ws + o; o += (size_t)BT_ * 50;

  auto gemm = [&](const float* A, const float* Bm, const float* bias, float* C,
                  int M, int N, int Kd,
                  long long lda, long long ldb, long long ldc,
                  int batches, int bdiv,
                  long long sA0, long long sA1, long long sB0, long long sB1,
                  long long sC0, long long sC1, int btrans, int act, float alpha) {
    dim3 grid((N + TN - 1) / TN, (M + TM - 1) / TM, batches);
    const bool regular = (M % TM == 0) && (N % TN == 0) && (Kd % TK == 0) &&
                         ((lda & 3) == 0) && ((ldb & 3) == 0);
    if (regular && !btrans)
      gemm_wmma_bf16<1><<<grid, 256, 0, stream>>>(A, Bm, bias, C, M, N, Kd,
                                                  lda, ldb, ldc, bdiv,
                                                  sA0, sA1, sB0, sB1, sC0, sC1,
                                                  btrans, act, alpha);
    else if (regular)
      gemm_wmma_bf16<2><<<grid, 256, 0, stream>>>(A, Bm, bias, C, M, N, Kd,
                                                  lda, ldb, ldc, bdiv,
                                                  sA0, sA1, sB0, sB1, sC0, sC1,
                                                  btrans, act, alpha);
    else
      gemm_wmma_bf16<0><<<grid, 256, 0, stream>>>(A, Bm, bias, C, M, N, Kd,
                                                  lda, ldb, ldc, bdiv,
                                                  sA0, sA1, sB0, sB1, sC0, sC1,
                                                  btrans, act, alpha);
  };

  const float qscale = powf((float)K_, -0.25f);

  // 1) char CNN + highway input
  char_cnn_kernel<<<BT_, 256, 0, stream>>>(char_emb, seqs,
      conv_w[0], conv_b[0], conv_w[1], conv_b[1],
      conv_w[2], conv_b[2], conv_w[3], conv_b[3], CE);
  build_hwin_kernel<<<(BT_ * HWD_ + 255) / 256, 256, 0, stream>>>(x, CE, HWIN);

  // 2) highway gates (WMMA GEMMs) + xcat assembly
  gemm(HWIN, hw_t1_w, hw_t1_b, F1, BT_, HWD_, HWD_, HWD_, HWD_, HWD_,
       1, 1, 0, 0, 0, 0, 0, 0, 0, 1, 1.f);                       // relu
  gemm(HWIN, hw_t2_w, hw_t2_b, TG, BT_, HWD_, HWD_, HWD_, HWD_, HWD_,
       1, 1, 0, 0, 0, 0, 0, 0, 0, 3, 1.f);                       // sigmoid
  build_xcat_kernel<<<(BT_ * EMB_ + 255) / 256, 256, 0, stream>>>(x, HWIN, F1, TG, XCAT);

  // 3) transformer blocks
  for (int blk = 0; blk < 4; ++blk) {
    const int base = (blk == 0) ? 0 : 15 + 13 * (blk - 1);
    const float* bu    = P(base + 0);
    const float* ff1_b = P(base + 1); const float* ff1_w = P(base + 2);
    const float* ff2_b = P(base + 3); const float* ff2_w = P(base + 4);
    const float* ln1_b = P(base + 5); const float* ln1_g = P(base + 6);
    const float* ln2_b = P(base + 7); const float* ln2_g = P(base + 8);
    const float* wk = P(base + (blk == 0 ? 11 : 9));
    const float* wq = P(base + (blk == 0 ? 12 : 10));
    const float* wu = P(base + (blk == 0 ? 13 : 11));
    const float* wv = P(base + (blk == 0 ? 14 : 12));

    const float* zin = (blk == 0) ? XCAT : Zb;
    const int e = (blk == 0) ? EMB_ : K_;

    // QKV projections (blocks 1-3 hit the fully-regular fast kernel)
    gemm(zin, wq, nullptr, Qb, BT_, HK_, e, e, HK_, HK_,
         1, 1, 0, 0, 0, 0, 0, 0, 0, 0, qscale);
    gemm(zin, wk, nullptr, Kb, BT_, HK_, e, e, HK_, HK_,
         1, 1, 0, 0, 0, 0, 0, 0, 0, 0, qscale);
    gemm(zin, wv, nullptr, Vb, BT_, HK_, e, e, HK_, HK_,
         1, 1, 0, 0, 0, 0, 0, 0, 0, 0, 1.f);

    // scores = Q @ K^T, batched over (b,h): 512 batches of 256x256x512
    gemm(Qb, Kb, nullptr, SC, T_, T_, K_, HK_, HK_, T_,
         B_ * H_, H_,
         (long long)T_ * HK_, K_,                      // A: b stride, h stride
         (long long)T_ * HK_, K_,                      // B
         (long long)H_ * T_ * T_, (long long)T_ * T_,  // C
         /*btrans=*/1, 0, 1.f);

    softmax_mask_kernel<<<B_ * H_ * T_, 256, 0, stream>>>(SC, pad);

    // context = att @ V
    gemm(SC, Vb, nullptr, AO, T_, K_, T_, T_, HK_, HK_,
         B_ * H_, H_,
         (long long)H_ * T_ * T_, (long long)T_ * T_,
         (long long)T_ * HK_, K_,
         (long long)T_ * HK_, K_,
         /*btrans=*/0, 0, 1.f);

    // unify heads, residual, LN, FFN
    gemm(AO, wu, bu, Ab, BT_, K_, HK_, HK_, K_, K_,
         1, 1, 0, 0, 0, 0, 0, 0, 0, 0, 1.f);
    const float* yptr;
    if (blk == 0) {
      const float* tr_b = P(9); const float* tr_w = P(10);
      gemm(XCAT, tr_w, tr_b, Yb, BT_, K_, EMB_, EMB_, K_, K_,
           1, 1, 0, 0, 0, 0, 0, 0, 0, 0, 1.f);
      yptr = Yb;
    } else {
      yptr = Zb;
    }
    add_ln_kernel<<<BT_, 256, 0, stream>>>(Ab, yptr, ln1_g, ln1_b, H1);
    gemm(H1, ff1_w, ff1_b, FF, BT_, 2048, K_, K_, 2048, 2048,
         1, 1, 0, 0, 0, 0, 0, 0, 0, 1, 1.f);                     // relu
    gemm(FF, ff2_w, ff2_b, Ab, BT_, K_, 2048, 2048, K_, K_,
         1, 1, 0, 0, 0, 0, 0, 0, 0, 0, 1.f);
    add_ln_kernel<<<BT_, 256, 0, stream>>>(Ab, H1, ln2_g, ln2_b, Zb);
  }

  // 4) head
  gemm(XCAT, transform_w, transform_b, PROJ, BT_, K_, EMB_, EMB_, K_, K_,
       1, 1, 0, 0, 0, 0, 0, 0, 0, 2, 1.f);                       // tanh
  mul_kernel<<<(BT_ * K_ + 255) / 256, 256, 0, stream>>>(PROJ, Zb, PZ, BT_ * K_);
  gemm(PZ, features_w, nullptr, FE, BT_, 50, K_, K_, 50, 50,
       1, 1, 0, 0, 0, 0, 0, 0, 0, 2, 1.f);                       // tanh

  // outputs: [log_probs (BT*2)] [z (BT*512)] [logits (BT*2)]
  float* out      = (float*)d_out;
  float* out_logp = out;
  float* out_z    = out + (size_t)BT_ * 2;
  float* out_lg   = out + (size_t)BT_ * 2 + (size_t)BT_ * K_;
  final_logits_kernel<<<(BT_ + 255) / 256, 256, 0, stream>>>(FE, toprobs_w, toprobs_b,
                                                             out_logp, out_lg);
  copy_kernel<<<(BT_ * K_ + 255) / 256, 256, 0, stream>>>(Zb, out_z, BT_ * K_);
}
